// UV_Aggregator_51092930953381
// MI455X (gfx1250) — matrix operand order repaired and checked by reading, not compiled, not run on hardware
//
#include <hip/hip_runtime.h>
#include <math.h>

// ---------------------------------------------------------------------------
// UV_Aggregator for MI455X (gfx1250): fp16 WMMA chain, wave32, 4 waves/node.
// ---------------------------------------------------------------------------

#define Dn 128      // embed dim
#define Hn 50       // history length
#define ROWS 64     // padded history rows (4 x 16 M-tiles)
#define ACT_STRIDE 264   // 264 ushorts = 528 B = 33*16 -> b128-aligned rows
#define O_STRIDE   136   // 136 ushorts = 272 B = 17*16

// ws layout (fp16/ushort element offsets): transposed weights Wt[nOut][K]
#define OFF_GE   0        // 128x128
#define OFF_WR1  16384    // 128x256
#define OFF_WR2  49152    // 128x128
#define OFF_ATT1 65536    // 128x256
#define OFF_ATT2 98304    // 128x128
// total 114688 ushorts = 229376 bytes of d_ws

typedef __attribute__((ext_vector_type(16))) _Float16 v16h;
typedef __attribute__((ext_vector_type(2)))  __fp16   v2fp16;
typedef __attribute__((ext_vector_type(8)))  float    v8f;

union HFrag {
    v16h v;
    uint4 q[2];
    unsigned int w[8];
    unsigned short u[16];
};

// Pack two fp32 -> fp16x2 with one v_cvt_pk_rtz_f16_f32.
__device__ __forceinline__ unsigned int f2h_pk(float lo, float hi) {
#if __has_builtin(__builtin_amdgcn_cvt_pkrtz)
    union { v2fp16 v; unsigned int u; } c;
    c.v = __builtin_amdgcn_cvt_pkrtz(lo, hi);
    return c.u;
#else
    union { _Float16 h[2]; unsigned int u; } c;
    c.h[0] = (_Float16)lo; c.h[1] = (_Float16)hi;
    return c.u;
#endif
}

__device__ __forceinline__ unsigned short f2h(float f) {
    union { _Float16 h; unsigned short u; } c;
    c.h = (_Float16)f;
    return c.u;
}
__device__ __forceinline__ float h2f(unsigned short b) {
    union { unsigned short u; _Float16 h; } c;
    c.u = b;
    return (float)c.h;
}

// A (16 x 32*KSTEPS, f16) from LDS, B = Wt[128][32*KSTEPS] (f16, global),
// 16x128 result via v_wmma_f32_16x16x32_f16; acc initialized by first WMMA
// with inline-0 C operand (no zero_acc mov burst).
template <int KSTEPS>
__device__ __forceinline__ void wmma_rowblock(const unsigned short* aBase, int aStride,
                                              const unsigned short* bBase,
                                              int lane, v8f acc[8]) {
    const int arow = lane & 15;
    const int ak   = (lane >> 4) * 8;    // A: K chunks [ak,ak+8) and [ak+16,ak+24)
    const int bcol = lane & 15;
    const int bk   = (lane >> 4) * 8;    // B: K chunks [bk,bk+8) and [bk+16,bk+24)
    const int bK   = KSTEPS * 32;

    const v8f zc = {0.0f, 0.0f, 0.0f, 0.0f, 0.0f, 0.0f, 0.0f, 0.0f};

    HFrag a[KSTEPS];
#pragma unroll
    for (int s = 0; s < KSTEPS; ++s) {
        const unsigned short* ap = aBase + arow * aStride + s * 32 + ak;
        a[s].q[0] = *(const uint4*)(ap);
        a[s].q[1] = *(const uint4*)(ap + 16);
    }
#pragma unroll
    for (int t = 0; t < 8; ++t) {
        const unsigned short* bp0 = bBase + (t * 16 + bcol) * bK + bk;
#pragma unroll
        for (int s = 0; s < KSTEPS; ++s) {
            HFrag b;
            const unsigned short* bp = bp0 + s * 32;
            b.q[0] = *(const uint4*)(bp);
            b.q[1] = *(const uint4*)(bp + 16);
            acc[t] = __builtin_amdgcn_wmma_f32_16x16x32_f16(
                false, a[s].v, false, b.v, (short)0,
                (s == 0) ? zc : acc[t], false, false);
        }
    }
}

// bias + relu + fp16 store of the 16x128 accumulator (C layout: VGPR i,
// lanes<16 -> M=i, N=lane; lanes>=16 -> M=8+i, N=lane-16).
// Rows i, i+1 share a column -> one packed cvt per pair.
__device__ __forceinline__ void store_act(v8f acc[8], const float* bias,
                                          unsigned short* dst, int dstStride,
                                          int lane,
                                          unsigned short* dst2, int dst2Stride) {
    const int col  = lane & 15;
    const int mrow = (lane >> 4) * 8;
#pragma unroll
    for (int t = 0; t < 8; ++t) {
        const int c = t * 16 + col;
        const float b = bias[c];
#pragma unroll
        for (int i = 0; i < 8; i += 2) {
            float v0 = fmaxf(acc[t][i] + b, 0.0f);
            float v1 = fmaxf(acc[t][i + 1] + b, 0.0f);
            unsigned int p = f2h_pk(v0, v1);
            unsigned short lo = (unsigned short)p;
            unsigned short hi = (unsigned short)(p >> 16);
            dst[(mrow + i) * dstStride + c]     = lo;
            dst[(mrow + i + 1) * dstStride + c] = hi;
            if (dst2) {
                dst2[(mrow + i) * dst2Stride + c]     = lo;
                dst2[(mrow + i + 1) * dst2Stride + c] = hi;
            }
        }
    }
}

// dst[n*K + k] = f16(src[k*N + n])   (transpose + fp32->fp16)
__global__ void transpose_cvt_kernel(const float* __restrict__ src,
                                     unsigned short* __restrict__ dst,
                                     int K, int N) {
    int i = blockIdx.x * blockDim.x + threadIdx.x;
    if (i >= K * N) return;
    int n = i / K, k = i - n * K;
    dst[i] = f2h(src[k * N + n]);
}

__global__ __launch_bounds__(128) void uvagg_kernel(
    const int* __restrict__ nodes, const int* __restrict__ history_uv,
    const int* __restrict__ history_r, const int* __restrict__ hist_len,
    const float* __restrict__ u2e, const float* __restrict__ v2e,
    const float* __restrict__ r2e,
    const float* __restrict__ ge_b, const float* __restrict__ wr1_b,
    const float* __restrict__ wr2_b, const float* __restrict__ att1_b,
    const float* __restrict__ att2_b, const float* __restrict__ att3_w,
    const float* __restrict__ att3_b,
    const unsigned short* __restrict__ wt,   // fp16 transposed weights in d_ws
    float* __restrict__ out) {

    __shared__ unsigned short s_act[ROWS * ACT_STRIDE];  // 33.0 KB
    __shared__ unsigned short s_O[ROWS * O_STRIDE];      // 17.0 KB
    __shared__ float s_logits[ROWS];

    const int node    = blockIdx.x;
    const int lane    = threadIdx.x & 31;
    const int wave    = threadIdx.x >> 5;
    const int rowbase = wave * 16;

    unsigned short* actW = s_act + rowbase * ACT_STRIDE;
    unsigned short* oW   = s_O   + rowbase * O_STRIDE;

    v8f acc[8];

    // ---------------- Stage 1: E = relu(u2e[hist_uv] @ ge_w + ge_b) ----------
    {
        const int arow = lane & 15;
        const int ak   = (lane >> 4) * 8;
        const int bcol = lane & 15;
        const int bk   = (lane >> 4) * 8;
        int gh = rowbase + arow;
        int hh = (gh < Hn) ? gh : (Hn - 1);
        int uidx = history_uv[node * Hn + hh];
        const float* up = u2e + (long)uidx * Dn;

        const v8f zc = {0.0f, 0.0f, 0.0f, 0.0f, 0.0f, 0.0f, 0.0f, 0.0f};

        HFrag a[4];
#pragma unroll
        for (int s = 0; s < 4; ++s) {
            const float* p = up + s * 32 + ak;
#pragma unroll
            for (int j = 0; j < 4; ++j) a[s].w[j]     = f2h_pk(p[2 * j], p[2 * j + 1]);
#pragma unroll
            for (int j = 0; j < 4; ++j) a[s].w[4 + j] = f2h_pk(p[16 + 2 * j], p[16 + 2 * j + 1]);
        }
        const unsigned short* bBase = wt + OFF_GE;
#pragma unroll
        for (int t = 0; t < 8; ++t) {
            const unsigned short* bp0 = bBase + (t * 16 + bcol) * 128 + bk;
#pragma unroll
            for (int s = 0; s < 4; ++s) {
                HFrag b;
                const unsigned short* bp = bp0 + s * 32;
                b.q[0] = *(const uint4*)(bp);
                b.q[1] = *(const uint4*)(bp + 16);
                acc[t] = __builtin_amdgcn_wmma_f32_16x16x32_f16(
                    false, a[s].v, false, b.v, (short)0,
                    (s == 0) ? zc : acc[t], false, false);
            }
        }
        store_act(acc, ge_b, actW, ACT_STRIDE, lane, nullptr, 0);
    }

    // ---- fill cols [128,256) with r2e rows (concat for wr1) ----------------
    {
        int r  = lane & 15;
        int gh = rowbase + r;
        int hh = (gh < Hn) ? gh : (Hn - 1);
        int ridx = history_r[node * Hn + hh];
        const float* rp = r2e + (long)ridx * Dn;
        int c0 = (lane >> 4) * 64;
        unsigned short* dst = actW + r * ACT_STRIDE + Dn + c0;
#pragma unroll 8
        for (int c = 0; c < 64; c += 2)
            *(unsigned int*)(dst + c) = f2h_pk(rp[c0 + c], rp[c0 + c + 1]);
    }

    // ---------------- Stage 2: X = relu([E|R] @ wr1 + b) --------------------
    wmma_rowblock<8>(actW, ACT_STRIDE, wt + OFF_WR1, lane, acc);
    store_act(acc, wr1_b, actW, ACT_STRIDE, lane, nullptr, 0);

    // ---------------- Stage 3: O = relu(X @ wr2 + b) ------------------------
    wmma_rowblock<4>(actW, ACT_STRIDE, wt + OFF_WR2, lane, acc);
    store_act(acc, wr2_b, actW, ACT_STRIDE, lane, oW, O_STRIDE);

    // ---- fill cols [128,256) with uv_rep = v2e[nodes[n]] (broadcast) -------
    {
        int r = lane & 15;
        const float* vp = v2e + (long)nodes[node] * Dn;
        int c0 = (lane >> 4) * 64;
        unsigned short* dst = actW + r * ACT_STRIDE + Dn + c0;
#pragma unroll 8
        for (int c = 0; c < 64; c += 2)
            *(unsigned int*)(dst + c) = f2h_pk(vp[c0 + c], vp[c0 + c + 1]);
    }

    // ---------------- Stage 4: A1 = relu([O|uv] @ att1 + b) -----------------
    wmma_rowblock<8>(actW, ACT_STRIDE, wt + OFF_ATT1, lane, acc);
    store_act(acc, att1_b, actW, ACT_STRIDE, lane, nullptr, 0);

    // ---------------- Stage 5: A2 = relu(A1 @ att2 + b) ---------------------
    wmma_rowblock<4>(actW, ACT_STRIDE, wt + OFF_ATT2, lane, acc);
    store_act(acc, att2_b, actW, ACT_STRIDE, lane, nullptr, 0);

    // ---------------- logits + mask -----------------------------------------
    if (lane < 16) {
        int gh = rowbase + lane;
        const unsigned short* ap = s_act + gh * ACT_STRIDE;
        float s = att3_b[0];
#pragma unroll 8
        for (int k = 0; k < Dn; ++k) s += h2f(ap[k]) * att3_w[k];
        int hl = hist_len[node];
        if (hl < 1) hl = 1;
        s_logits[gh] = (gh < hl) ? s : -INFINITY;
    }
    __syncthreads();

    // ---------------- softmax over H + weighted sum of O --------------------
    {
        int d = threadIdx.x;   // 0..127
        float m = -INFINITY;
#pragma unroll 8
        for (int h = 0; h < ROWS; ++h) m = fmaxf(m, s_logits[h]);
        float denom = 0.0f, accum = 0.0f;
#pragma unroll 4
        for (int h = 0; h < ROWS; ++h) {
            float e = __expf(s_logits[h] - m);
            denom += e;
            accum += e * h2f(s_O[h * O_STRIDE + d]);
        }
        out[(long)node * Dn + d] = accum / denom;
    }
}

extern "C" void kernel_launch(void* const* d_in, const int* in_sizes, int n_in,
                              void* d_out, int out_size, void* d_ws, size_t ws_size,
                              hipStream_t stream) {
    const int*   nodes  = (const int*)d_in[0];
    const int*   huv    = (const int*)d_in[1];
    const int*   hr     = (const int*)d_in[2];
    const int*   hlen   = (const int*)d_in[3];
    const float* u2e    = (const float*)d_in[4];
    const float* v2e    = (const float*)d_in[5];
    const float* r2e    = (const float*)d_in[6];
    const float* ge_w   = (const float*)d_in[7];
    const float* ge_b   = (const float*)d_in[8];
    const float* wr1_w  = (const float*)d_in[9];
    const float* wr1_b  = (const float*)d_in[10];
    const float* wr2_w  = (const float*)d_in[11];
    const float* wr2_b  = (const float*)d_in[12];
    const float* att1_w = (const float*)d_in[13];
    const float* att1_b = (const float*)d_in[14];
    const float* att2_w = (const float*)d_in[15];
    const float* att2_b = (const float*)d_in[16];
    const float* att3_w = (const float*)d_in[17];
    const float* att3_b = (const float*)d_in[18];
    float* out = (float*)d_out;
    unsigned short* ws = (unsigned short*)d_ws;  // needs 229376 bytes

    const int N = in_sizes[0];  // 4096 nodes

    // Transpose + fp16-convert the five weight matrices into d_ws.
    auto launchT = [&](const float* src, unsigned short* dst, int K, int Nc) {
        int total = K * Nc;
        transpose_cvt_kernel<<<(total + 255) / 256, 256, 0, stream>>>(src, dst, K, Nc);
    };
    launchT(ge_w,   ws + OFF_GE,   128, 128);
    launchT(wr1_w,  ws + OFF_WR1,  256, 128);
    launchT(wr2_w,  ws + OFF_WR2,  128, 128);
    launchT(att1_w, ws + OFF_ATT1, 256, 128);
    launchT(att2_w, ws + OFF_ATT2, 128, 128);

    uvagg_kernel<<<N, 128, 0, stream>>>(nodes, huv, hr, hlen, u2e, v2e, r2e,
                                        ge_b, wr1_b, wr2_b, att1_b, att2_b,
                                        att3_w, att3_b, ws, out);
}
